// GraphormerAttentionHead_2216203124824
// MI455X (gfx1250) — matrix operand order repaired and checked by reading, not compiled, not run on hardware
//
#include <hip/hip_runtime.h>
#include <math.h>

// ---------------- problem constants ----------------
#define NN    6144        // total nodes
#define DIMI  128         // input dim
#define DIMQ  64          // q/k/v dim
#define EDIM  16          // edge feature dim
#define LMAX  5           // max path length
#define PNUM  786432      // path pairs
#define GNUM  48          // graphs (128 nodes each)

typedef __attribute__((ext_vector_type(16))) __bf16 v16bf;
typedef __attribute__((ext_vector_type(8)))  __bf16 v8bf;
typedef __attribute__((ext_vector_type(8)))  float  v8f;
typedef __attribute__((ext_vector_type(4)))  unsigned v4u;
typedef __attribute__((ext_vector_type(8)))  int      v8i;
typedef __attribute__((ext_vector_type(4)))  int      v4i;

// f32 -> bf16 (round-to-nearest-even via bit trick)
static __device__ inline __bf16 tobf(float f) {
    unsigned u = __builtin_bit_cast(unsigned, f);
    u += 0x7FFFu + ((u >> 16) & 1u);
    unsigned short h = (unsigned short)(u >> 16);
    return __builtin_bit_cast(__bf16, h);
}

// A-matrix fragment (16x32 bf16): lane holds row (lane&15); element e<8 -> K=base+e,
// element e>=8 -> K=base+16+(e-8); base includes 8*(lane>>4).
static __device__ inline v16bf fragA_bf(const __bf16* row, int base) {
    v8bf lo = *(const v8bf*)(row + base);
    v8bf hi = *(const v8bf*)(row + base + 16);
    v16bf r;
#pragma unroll
    for (int i = 0; i < 8; ++i) { r[i] = lo[i]; r[i + 8] = hi[i]; }
    return r;
}
static __device__ inline v16bf fragA_f32(const float* row, int base) {
    v16bf r;
#pragma unroll
    for (int i = 0; i < 8; ++i) { r[i] = tobf(row[base + i]); r[i + 8] = tobf(row[base + 16 + i]); }
    return r;
}
// B-matrix fragment (32x16 bf16): lane holds column (lane&15); element e<8 -> K=base+e,
// element e>=8 -> K=base+16+(e-8); base includes 8*(lane>>4).
static __device__ inline v16bf fragB_bf(const __bf16* row, int base) {
    v8bf lo = *(const v8bf*)(row + base);
    v8bf hi = *(const v8bf*)(row + base + 16);
    v16bf r;
#pragma unroll
    for (int i = 0; i < 8; ++i) { r[i] = lo[i]; r[i + 8] = hi[i]; }
    return r;
}
static __device__ inline v16bf fragB_f32(const float* row, int base) {
    v16bf r;
#pragma unroll
    for (int i = 0; i < 8; ++i) { r[i] = tobf(row[base + i]); r[i + 8] = tobf(row[base + 16 + i]); }
    return r;
}

// ---------------- Tensor Data Mover: 2D tile (32 rows x 128 f32) -> LDS ----------------
#if defined(__gfx1250__) && __has_builtin(__builtin_amdgcn_tensor_load_to_lds)
#define HAVE_TDM 1
#endif

// DMA a 32x128 f32 tile (row stride NN) from global into LDS at byte addr lds_addr.
static __device__ inline void tdm_tile_load(unsigned lds_addr, const float* gsrc) {
#ifdef HAVE_TDM
    const unsigned long long ga = (unsigned long long)(uintptr_t)gsrc;
    v4u g0;
    g0[0] = 1u;                                              // count=1, user-mode descriptor
    g0[1] = lds_addr;                                        // LDS byte address
    g0[2] = (unsigned)(ga & 0xFFFFFFFFu);                    // global_addr[31:0]
    g0[3] = (unsigned)((ga >> 32) & 0x1FFFFFFu) | (2u << 30);// global_addr[56:32] | type=2
    v8i g1;
    g1[0] = (int)(2u << 16);            // workgroup_mask=0, data_size=2 (4B)
    g1[1] = (int)(6144u << 16);         // tensor_dim0[15:0]=6144 (atomic_barrier_addr=0)
    g1[2] = (int)(6144u << 16);         // tensor_dim0[31:16]=0, tensor_dim1[15:0]=6144
    g1[3] = (int)(128u << 16);          // tensor_dim1[31:16]=0, tile_dim0=128
    g1[4] = 32;                         // tile_dim1=32, tile_dim2=0
    g1[5] = 6144;                       // tensor_dim0_stride[31:0]
    g1[6] = 0;                          // stride[47:32]=0, tensor_dim1_stride lo=0
    g1[7] = 0;
    v4i g2 = {0, 0, 0, 0};
    v4i g3 = {0, 0, 0, 0};
    v8i g4 = {0, 0, 0, 0, 0, 0, 0, 0};  // extra operand of 6-arg toolchain variant
    __builtin_amdgcn_tensor_load_to_lds(g0, g1, g2, g3, g4, 0);
#else
    (void)lds_addr; (void)gsrc;
#endif
}

static __device__ inline void tdm_tile_load_fallback(float* dst, const float* gsrc, int lane) {
#ifndef HAVE_TDM
    for (int i = lane; i < 32 * 128; i += 32)
        dst[i] = gsrc[(i >> 7) * NN + (i & 127)];
#else
    (void)dst; (void)gsrc; (void)lane;
#endif
}

static __device__ inline void tdm_wait() {
#ifdef HAVE_TDM
    __builtin_amdgcn_s_wait_tensorcnt(0);
#endif
}

// ---------------- kernel 1: fused q/k/v projections ----------------
__global__ __launch_bounds__(256) void qkv_proj_kernel(
    const float* __restrict__ Xq, const float* __restrict__ Xk, const float* __restrict__ Xv,
    const float* __restrict__ Wq, const float* __restrict__ bq,
    const float* __restrict__ Wk, const float* __restrict__ bk,
    const float* __restrict__ Wv, const float* __restrict__ bv,
    __bf16* __restrict__ qbf, __bf16* __restrict__ kbf, __bf16* __restrict__ vT)
{
    const int z = blockIdx.z;
    const float* X = (z == 0) ? Xq : (z == 1) ? Xk : Xv;
    const float* W = (z == 0) ? Wq : (z == 1) ? Wk : Wv;
    const float* B = (z == 0) ? bq : (z == 1) ? bk : bv;
    __bf16* outRM  = (z == 0) ? qbf : kbf;

    const int w = threadIdx.x >> 5, lane = threadIdx.x & 31;
    const int laneLo = lane & 15, laneHi = lane >> 4;
    const int rowbase = blockIdx.x * 128 + w * 16;

    const float* xrow = X + (rowbase + laneLo) * DIMI;
    v16bf a[4];
#pragma unroll
    for (int kk = 0; kk < 4; ++kk) a[kk] = fragA_f32(xrow, kk * 32 + 8 * laneHi);

#pragma unroll
    for (int cg = 0; cg < 4; ++cg) {
        const float* wrow = W + (cg * 16 + laneLo) * DIMI;
        v8f acc = {0.f, 0.f, 0.f, 0.f, 0.f, 0.f, 0.f, 0.f};
#pragma unroll
        for (int kk = 0; kk < 4; ++kk) {
            v16bf bf = fragB_f32(wrow, kk * 32 + 8 * laneHi);
            acc = __builtin_amdgcn_wmma_f32_16x16x32_bf16(false, a[kk], false, bf,
                                                          (short)0, acc, false, false);
        }
        const float bias = B[cg * 16 + laneLo];
#pragma unroll
        for (int v = 0; v < 8; ++v) {
            const int m = v + 8 * laneHi;
            const int node = rowbase + m;
            const __bf16 h = tobf(acc[v] + bias);
            if (z == 2) vT[(cg * 16 + laneLo) * NN + node] = h;
            else        outRM[node * DIMQ + cg * 16 + laneLo] = h;
        }
    }
}

// ---------------- kernel 2: zero the c matrix ----------------
__global__ void zero_c_kernel(float4* __restrict__ p, int n4) {
    const float4 z = make_float4(0.f, 0.f, 0.f, 0.f);
    for (int i = blockIdx.x * blockDim.x + threadIdx.x; i < n4; i += gridDim.x * blockDim.x)
        p[i] = z;
}

// ---------------- kernel 3: edge-encoding scatter into c ----------------
__global__ __launch_bounds__(256) void edge_enc_kernel(
    const float* __restrict__ edge_attr, const float* __restrict__ edge_vector,
    const int* __restrict__ path_src, const int* __restrict__ path_dst,
    const int* __restrict__ path_edges, const int* __restrict__ path_len,
    float* __restrict__ cmat)
{
    __shared__ float ev[LMAX * EDIM];
    if (threadIdx.x < LMAX * EDIM) ev[threadIdx.x] = edge_vector[threadIdx.x];
    __syncthreads();

    const int p = blockIdx.x * blockDim.x + threadIdx.x;
    if (p >= PNUM) return;
    const int len = path_len[p];
    float acc = 0.f;
#pragma unroll
    for (int l = 0; l < LMAX; ++l) {
        if (l < len) {
            const int e = path_edges[p * LMAX + l];
            const float4* ea = (const float4*)(edge_attr + (size_t)e * EDIM);
            const float* evl = ev + l * EDIM;
            float s = 0.f;
#pragma unroll
            for (int q4 = 0; q4 < 4; ++q4) {
                float4 x = ea[q4];
                s += x.x * evl[q4 * 4 + 0] + x.y * evl[q4 * 4 + 1]
                   + x.z * evl[q4 * 4 + 2] + x.w * evl[q4 * 4 + 3];
            }
            acc += s;
        }
    }
    cmat[(size_t)path_src[p] * NN + path_dst[p]] = acc / (float)len;
}

// ---------------- kernel 4: block-diagonal attention, TDM-staged b+c ----------------
// 64 threads = 2 waves; block owns 32 rows. Dynamic LDS:
//   [0,32768)    : bbuf double buffer, 2 x (32x128) f32
//   [32768,65536): cbuf double buffer
//   [65536,73728): p staging, 2 x (16x128) bf16
__global__ __launch_bounds__(64) void attn_kernel(
    const __bf16* __restrict__ qbf, const __bf16* __restrict__ kbf,
    const __bf16* __restrict__ vT,  const float* __restrict__ bmat,
    const float* __restrict__ cmat, float* __restrict__ out)
{
    extern __shared__ __align__(16) char smem[];
    float*  bbuf = (float*)smem;
    float*  cbuf = (float*)(smem + 32768);
    __bf16* plds = (__bf16*)(smem + 65536);

    const int tid = threadIdx.x;
    const int w = tid >> 5, lane = tid & 31;
    const int laneLo = lane & 15, laneHi = lane >> 4;
    const int rowbase32 = blockIdx.x * 32;
    const int rowbase = rowbase32 + w * 16;
    const int g = rowbase32 >> 7;                 // graph / diagonal tile index
    const float scale = 0.125f;                   // 1/sqrt(64)

    // Q fragments for this stripe (K = 0..63)
    const __bf16* qrow = qbf + (rowbase + laneLo) * DIMQ;
    const v16bf aq0 = fragA_bf(qrow, 8 * laneHi);
    const v16bf aq1 = fragA_bf(qrow, 32 + 8 * laneHi);

    float M[8], D[8];
#pragma unroll
    for (int v = 0; v < 8; ++v) { M[v] = -INFINITY; D[v] = 0.f; }

    // preload tile 0 (b and c) via TDM into buffer 0
    if (w == 0) {
        tdm_tile_load((unsigned)(uintptr_t)(bbuf), bmat + (size_t)rowbase32 * NN);
        tdm_tile_load((unsigned)(uintptr_t)(cbuf), cmat + (size_t)rowbase32 * NN);
        tdm_tile_load_fallback(bbuf, bmat + (size_t)rowbase32 * NN, lane);
        tdm_tile_load_fallback(cbuf, cmat + (size_t)rowbase32 * NN, lane);
    }

    // ---- phase 1: running row max & softmax denominator over ALL 48 col tiles ----
    for (int ct = 0; ct < GNUM; ++ct) {
        if (w == 0) tdm_wait();
        __syncthreads();                          // tile `ct` resident; prior reads done
        const int buf = ct & 1;
        if (w == 0 && ct + 1 < GNUM) {            // DMA next tile into the other buffer
            const float* gb = bmat + (size_t)rowbase32 * NN + (ct + 1) * 128;
            const float* gc = cmat + (size_t)rowbase32 * NN + (ct + 1) * 128;
            tdm_tile_load((unsigned)(uintptr_t)(bbuf + (buf ^ 1) * 4096), gb);
            tdm_tile_load((unsigned)(uintptr_t)(cbuf + (buf ^ 1) * 4096), gc);
            tdm_tile_load_fallback(bbuf + (buf ^ 1) * 4096, gb, lane);
            tdm_tile_load_fallback(cbuf + (buf ^ 1) * 4096, gc, lane);
        }
        if (ct + 1 < GNUM)                        // prefetch next K tile rows into L2/L1
            __builtin_prefetch(kbf + (size_t)((ct + 1) * 128 + lane * 4) * DIMQ, 0, 0);

        const float* bt = bbuf + buf * 4096;
        const float* ctl = cbuf + buf * 4096;
        const float maskmul = (ct == g) ? 1.f : -1e6f;
        float s[8][8];                            // [cg][v]
#pragma unroll
        for (int cg = 0; cg < 8; ++cg) {
            const __bf16* krow = kbf + (ct * 128 + cg * 16 + laneLo) * DIMQ;
            v16bf b0 = fragB_bf(krow, 8 * laneHi);
            v16bf b1 = fragB_bf(krow, 32 + 8 * laneHi);
            v8f acc = {0.f, 0.f, 0.f, 0.f, 0.f, 0.f, 0.f, 0.f};
            acc = __builtin_amdgcn_wmma_f32_16x16x32_bf16(false, aq0, false, b0, (short)0, acc, false, false);
            acc = __builtin_amdgcn_wmma_f32_16x16x32_bf16(false, aq1, false, b1, (short)0, acc, false, false);
            const int lc = cg * 16 + laneLo;
#pragma unroll
            for (int v = 0; v < 8; ++v) {
                const int lr = w * 16 + v + 8 * laneHi;
                const float bc = bt[lr * 128 + lc] + ctl[lr * 128 + lc];
                s[cg][v] = (acc[v] * scale + bc) * maskmul;
            }
        }
        // per-row (cross-lane within 16-lane half) max + exp-sum, online update
#pragma unroll
        for (int v = 0; v < 8; ++v) {
            float lm = s[0][v];
#pragma unroll
            for (int cg = 1; cg < 8; ++cg) lm = fmaxf(lm, s[cg][v]);
#pragma unroll
            for (int off = 1; off < 16; off <<= 1) lm = fmaxf(lm, __shfl_xor(lm, off, 32));
            const float nM = fmaxf(M[v], lm);
            float le = 0.f;
#pragma unroll
            for (int cg = 0; cg < 8; ++cg) le += __expf(s[cg][v] - nM);
#pragma unroll
            for (int off = 1; off < 16; off <<= 1) le += __shfl_xor(le, off, 32);
            D[v] = D[v] * __expf(M[v] - nM) + le;
            M[v] = nM;
        }
    }

    // ---- reload diagonal b+c tile (L2-resident) into buffer 0 via TDM ----
    if (w == 0) {
        const float* gb = bmat + (size_t)rowbase32 * NN + g * 128;
        const float* gc = cmat + (size_t)rowbase32 * NN + g * 128;
        tdm_tile_load((unsigned)(uintptr_t)(bbuf), gb);
        tdm_tile_load((unsigned)(uintptr_t)(cbuf), gc);
        tdm_tile_load_fallback(bbuf, gb, lane);
        tdm_tile_load_fallback(cbuf, gc, lane);
        tdm_wait();
    }
    __syncthreads();

    // ---- phase 2: recompute diagonal tile, p = exp(s - M), stage to LDS ----
#pragma unroll
    for (int cg = 0; cg < 8; ++cg) {
        const __bf16* krow = kbf + (g * 128 + cg * 16 + laneLo) * DIMQ;
        v16bf b0 = fragB_bf(krow, 8 * laneHi);
        v16bf b1 = fragB_bf(krow, 32 + 8 * laneHi);
        v8f acc = {0.f, 0.f, 0.f, 0.f, 0.f, 0.f, 0.f, 0.f};
        acc = __builtin_amdgcn_wmma_f32_16x16x32_bf16(false, aq0, false, b0, (short)0, acc, false, false);
        acc = __builtin_amdgcn_wmma_f32_16x16x32_bf16(false, aq1, false, b1, (short)0, acc, false, false);
        const int lc = cg * 16 + laneLo;
#pragma unroll
        for (int v = 0; v < 8; ++v) {
            const int m = v + 8 * laneHi;
            const int lr = w * 16 + m;
            const float bc = bbuf[lr * 128 + lc] + cbuf[lr * 128 + lc];
            const float sv = acc[v] * scale + bc;   // diagonal: mask = 1
            plds[(w * 16 + m) * 128 + lc] = tobf(__expf(sv - M[v]));
        }
    }
    __syncthreads();

    // ---- P (16x128) @ V_block (128x64) via WMMA; V read from transposed bf16 ----
    const __bf16* prow = plds + (w * 16 + laneLo) * 128;
    v16bf ap[4];
#pragma unroll
    for (int kk = 0; kk < 4; ++kk) ap[kk] = fragA_bf(prow, kk * 32 + 8 * laneHi);

#pragma unroll
    for (int cf = 0; cf < 4; ++cf) {
        const __bf16* vrow = vT + (size_t)(cf * 16 + laneLo) * NN + g * 128;
        v8f acc = {0.f, 0.f, 0.f, 0.f, 0.f, 0.f, 0.f, 0.f};
#pragma unroll
        for (int kk = 0; kk < 4; ++kk) {
            v16bf bv = fragB_bf(vrow, kk * 32 + 8 * laneHi);
            acc = __builtin_amdgcn_wmma_f32_16x16x32_bf16(false, ap[kk], false, bv, (short)0, acc, false, false);
        }
#pragma unroll
        for (int v = 0; v < 8; ++v) {
            const int m = v + 8 * laneHi;
            out[(rowbase + m) * DIMQ + cf * 16 + laneLo] = acc[v] / D[v];
        }
    }
}

// ---------------- host-side launch ----------------
extern "C" void kernel_launch(void* const* d_in, const int* in_sizes, int n_in,
                              void* d_out, int out_size, void* d_ws, size_t ws_size,
                              hipStream_t stream)
{
    const float* query       = (const float*)d_in[0];
    const float* key         = (const float*)d_in[1];
    const float* value       = (const float*)d_in[2];
    const float* edge_attr   = (const float*)d_in[3];
    const float* bmat        = (const float*)d_in[4];
    const float* Wq          = (const float*)d_in[5];
    const float* bq          = (const float*)d_in[6];
    const float* Wk          = (const float*)d_in[7];
    const float* bk          = (const float*)d_in[8];
    const float* Wv          = (const float*)d_in[9];
    const float* bv          = (const float*)d_in[10];
    const float* edge_vector = (const float*)d_in[11];
    const int*   path_src    = (const int*)d_in[12];
    const int*   path_dst    = (const int*)d_in[13];
    const int*   path_edges  = (const int*)d_in[14];
    const int*   path_len    = (const int*)d_in[15];
    float* out = (float*)d_out;

    // workspace layout
    const size_t QK_BYTES = (size_t)NN * DIMQ * 2;   // 786432 B per bf16 matrix
    char* ws = (char*)d_ws;
    __bf16* qbf = (__bf16*)(ws);
    __bf16* kbf = (__bf16*)(ws + QK_BYTES);
    __bf16* vT  = (__bf16*)(ws + 2 * QK_BYTES);
    float*  cmat = (float*)(ws + 3 * QK_BYTES);       // NN*NN f32

    // 1) q/k/v projections (WMMA bf16)
    qkv_proj_kernel<<<dim3(NN / 128, 1, 3), 256, 0, stream>>>(
        query, key, value, Wq, bq, Wk, bk, Wv, bv, qbf, kbf, vT);

    // 2) zero c
    const int n4 = (NN * (NN / 4));
    zero_c_kernel<<<4096, 256, 0, stream>>>((float4*)cmat, n4);

    // 3) edge-encoding scatter
    edge_enc_kernel<<<(PNUM + 255) / 256, 256, 0, stream>>>(
        edge_attr, edge_vector, path_src, path_dst, path_edges, path_len, cmat);

    // 4) attention (dynamic LDS: 64KB b/c double buffers + 8KB p staging)
    attn_kernel<<<NN / 32, 64, 73728, stream>>>(qbf, kbf, vT, bmat, cmat, out);
}
